// BonsaiLayer_2078764171322
// MI455X (gfx1250) — matrix operand order repaired and checked by reading, not compiled, not run on hardware
//
#include <hip/hip_runtime.h>
#include <math.h>

// Problem constants (from reference)
#define D_DIM   1024
#define P_DIM   64
#define B_DIM   65536
#define C_DIM   10
#define INT_N   31      // internal tree nodes (th rows)
#define TOT_N   63      // total tree nodes (probs rows)
#define ROWS    630     // C * TOT
#define NW      4       // waves per workgroup (wave32)
#define COLS_WG 64      // batch rows per workgroup (16 per wave)
#define LDPF    68      // padded f32 LDS stride for Xp^T (68*4 = 272 B, 16B-aligned)

// Pre-packed fragment arrays in workspace (uint = bf16x2, 256 uints per fragment)
#define ZF_FRAGS   (32 * 4)          // kk-major: f = kk*4 + ns
#define WF_FRAGS   (40 * 2)          // f = t*2 + kk
#define TF_FRAGS   (2 * 2)
#define ZF_OFF     0
#define WF_OFF     (ZF_OFF + ZF_FRAGS * 256)
#define VF_OFF     (WF_OFF + WF_FRAGS * 256)
#define TF_OFF     (VF_OFF + WF_FRAGS * 256)
#define WS_UINTS   (TF_OFF + TF_FRAGS * 256)

typedef __attribute__((ext_vector_type(16))) __bf16        v16bf;
typedef __attribute__((ext_vector_type(8)))  float         v8f;
typedef __attribute__((ext_vector_type(4)))  float         v4f;
typedef __attribute__((ext_vector_type(8)))  unsigned int  v8u;
typedef __attribute__((ext_vector_type(4)))  unsigned int  v4u;

// Pack two f32 into bf16x2 (lo=a, hi=b), round-half-up: one v_perm_b32.
__device__ __forceinline__ unsigned pack_bf16x2(float a, float b) {
    unsigned ua = __builtin_bit_cast(unsigned, a) + 0x8000u;
    unsigned ub = __builtin_bit_cast(unsigned, b) + 0x8000u;
    return __builtin_amdgcn_perm(ub, ua, 0x07060302u);
}

__device__ __forceinline__ v8f wmma_bf16(v16bf a, v16bf b, v8f c) {
    return __builtin_amdgcn_wmma_f32_16x16x32_bf16(
        false, a, false, b, (short)0, c, false, false);
}

// tanh(1e9*x) in f32 == sign(x) (saturates past 1 ulp of 1.0 for any |x|>2e-8).
__device__ __forceinline__ float sign_sat(float x) {
    return (x > 0.0f) ? 1.0f : ((x < 0.0f) ? -1.0f : 0.0f);
}

// Load a pre-packed fragment: 2x global_load_b128, no conversion ALU.
__device__ __forceinline__ v16bf frag_pre(const unsigned* __restrict__ F, int fidx) {
    const int lane = threadIdx.x & 31;
    const v4u* p = (const v4u*)(F + ((size_t)fidx * 32 + lane) * 8);
    v4u lo = p[0], hi = p[1];
    v8u u;
    u[0] = lo[0]; u[1] = lo[1]; u[2] = lo[2]; u[3] = lo[3];
    u[4] = hi[0]; u[5] = hi[1]; u[6] = hi[2]; u[7] = hi[3];
    return __builtin_bit_cast(v16bf, u);
}

// A-fragment (16x32 bf16) of streamed X: row = brow0 + lane&15 (always in range).
// A layout: slot v -> K = half*16 + 2v ; one 16-float run.
__device__ __forceinline__ v16bf frag_a_x(const float* __restrict__ X,
                                          int brow0, int k0) {
    const int lane = threadIdx.x & 31;
    const int half = lane >> 4;
    const float* src = X + (size_t)(brow0 + (lane & 15)) * D_DIM + k0 + (half << 4);
    v4f g0a = *(const v4f*)(src +  0);
    v4f g0b = *(const v4f*)(src +  4);
    v4f g1a = *(const v4f*)(src +  8);
    v4f g1b = *(const v4f*)(src + 12);
    v8u u;
    u[0] = pack_bf16x2(g0a[0], g0a[1]);  u[1] = pack_bf16x2(g0a[2], g0a[3]);
    u[2] = pack_bf16x2(g0b[0], g0b[1]);  u[3] = pack_bf16x2(g0b[2], g0b[3]);
    u[4] = pack_bf16x2(g1a[0], g1a[1]);  u[5] = pack_bf16x2(g1a[2], g1a[3]);
    u[6] = pack_bf16x2(g1b[0], g1b[1]);  u[7] = pack_bf16x2(g1b[2], g1b[3]);
    return __builtin_bit_cast(v16bf, u);
}

// B-fragment of Xp from f32 LDS (Xp^T b-major: xp[b*LDPF + p]).
// B layout: slot v -> K = half*16 + 2v ; 16 contiguous f32 -> 4x ds_load_b128.
__device__ __forceinline__ v16bf frag_b_xpf(const float* xp, int bbase, int k0) {
    const int lane = threadIdx.x & 31;
    const v4f* s = (const v4f*)(xp + (size_t)(bbase + (lane & 15)) * LDPF
                                   + k0 + ((lane >> 4) << 4));
    v4f a0 = s[0], a1 = s[1], a2 = s[2], a3 = s[3];
    v8u u;
    u[0] = pack_bf16x2(a0[0], a0[1]);  u[1] = pack_bf16x2(a0[2], a0[3]);
    u[2] = pack_bf16x2(a1[0], a1[1]);  u[3] = pack_bf16x2(a1[2], a1[3]);
    u[4] = pack_bf16x2(a2[0], a2[1]);  u[5] = pack_bf16x2(a2[2], a2[3]);
    u[6] = pack_bf16x2(a3[0], a3[1]);  u[7] = pack_bf16x2(a3[2], a3[3]);
    return __builtin_bit_cast(v16bf, u);
}

// ---------------- Pre-pack kernels (run once per launch, tiny) ----------------
// A-layout frags: f = t*nkk + kk ; one thread per packed uint.
__global__ void pack_a_frags(const float* __restrict__ src, unsigned* __restrict__ dst,
                             int nrows, int ld, int ntiles, int nkk) {
    int idx = blockIdx.x * blockDim.x + threadIdx.x;
    if (idx >= ntiles * nkk * 256) return;
    int v    = idx & 7;
    int lane = (idx >> 3) & 31;
    int f    = idx >> 8;
    int kk   = f % nkk;
    int t    = f / nkk;
    int row  = t * 16 + (lane & 15);
    int half = lane >> 4;
    int k    = kk * 32 + (half << 4) + (v << 1);
    float x0 = 0.0f, x1 = 0.0f;
    if (row < nrows) {
        x0 = src[(size_t)row * ld + k];
        x1 = src[(size_t)row * ld + k + 1];
    }
    dst[idx] = pack_bf16x2(x0, x1);
}

// B-layout frags of src^T (B(k,n) = src[n][k]): f = kk*nns + ns.
__global__ void pack_b_frags(const float* __restrict__ src, unsigned* __restrict__ dst,
                             int ld, int nns, int nkk) {
    int idx = blockIdx.x * blockDim.x + threadIdx.x;
    if (idx >= nns * nkk * 256) return;
    int v    = idx & 7;
    int lane = (idx >> 3) & 31;
    int f    = idx >> 8;
    int ns   = f % nns;
    int kk   = f / nns;
    int n    = ns * 16 + (lane & 15);
    int half = lane >> 4;
    int k    = kk * 32 + (half << 4) + (v << 1);
    const float* s = src + (size_t)n * ld + k;
    dst[idx] = pack_bf16x2(s[0], s[1]);
}

// ---------------- Fused main kernel ----------------
__global__ __launch_bounds__(32 * NW)
void bonsai_fused_kernel(const float* __restrict__ X,
                         const unsigned* __restrict__ Zf,
                         const unsigned* __restrict__ Wf,
                         const unsigned* __restrict__ Vf,
                         const unsigned* __restrict__ Tf,
                         float* __restrict__ out) {
    __shared__ float xpf[COLS_WG * LDPF];           // Xp^T tile, f32 (17.4 KB)
    __shared__ float thv [NW][INT_N][16];
    __shared__ float prb [NW][TOT_N][16];
    __shared__ float sco [NW][C_DIM][16];

    const int wave = threadIdx.x >> 5;
    const int lane = threadIdx.x & 31;
    const int half = lane >> 4;
    const int ncol = lane & 15;
    const int b0   = blockIdx.x * COLS_WG;

    // ---- Stage 1: Xp^T = X @ Z^T / P ; X is the A operand (streamed once) ----
    {
        v8f acc[4] = {};
        for (int kk = 0; kk < 32; ++kk) {
            v16bf a = frag_a_x(X, b0 + wave * 16, kk * 32);
            __builtin_prefetch(X + (size_t)(b0 + wave * 16 + (lane & 15)) * D_DIM
                                 + kk * 32 + 64, 0, 3);
#pragma unroll
            for (int ns = 0; ns < 4; ++ns)
                acc[ns] = wmma_bf16(a, frag_pre(Zf, kk * 4 + ns), acc[ns]);
        }
        // D tile: element r -> (b = wave*16 + r + half*8, p = ns*16 + ncol)
#pragma unroll
        for (int ns = 0; ns < 4; ++ns) {
#pragma unroll
            for (int r = 0; r < 8; ++r) {
                int bl = wave * 16 + r + half * 8;
                xpf[(size_t)bl * LDPF + ns * 16 + ncol] = acc[ns][r] * 0.015625f;
            }
        }
    }
    __syncthreads();

    // Resident Xp B-fragments for this wave's 16 batch rows (K=64 -> two steps)
    const v16bf xb0 = frag_b_xpf(xpf, wave * 16, 0);
    const v16bf xb1 = frag_b_xpf(xpf, wave * 16, 32);

    // ---- Stage 2a: th = tanh(1e9 * T @ Xp) == sign(T @ Xp) in f32 ----
    {
        v8f t0 = {}, t1 = {};
        t0 = wmma_bf16(frag_pre(Tf, 0), xb0, t0);
        t0 = wmma_bf16(frag_pre(Tf, 1), xb1, t0);
        t1 = wmma_bf16(frag_pre(Tf, 2), xb0, t1);
        t1 = wmma_bf16(frag_pre(Tf, 3), xb1, t1);
#pragma unroll
        for (int r = 0; r < 8; ++r) {
            int row0 = r + half * 8;
            thv[wave][row0][ncol] = sign_sat(t0[r]);
            int row1 = 16 + r + half * 8;
            if (row1 < INT_N) thv[wave][row1][ncol] = sign_sat(t1[r]);
        }
    }
    if (lane < 16) {
#pragma unroll
        for (int c = 0; c < C_DIM; ++c) sco[wave][c][lane] = 0.0f;
    }
    __syncthreads();

    // ---- Stage 2b: tree path probabilities (heap order) ----
    if (lane < 16) {
        prb[wave][0][lane] = 1.0f;
        for (int n = 0; n < INT_N; ++n) {
            float p = prb[wave][n][lane];
            float t = thv[wave][n][lane];
            prb[wave][2 * n + 1][lane] = p * (1.0f + t) * 0.5f;
            prb[wave][2 * n + 2][lane] = p * (1.0f - t) * 0.5f;
        }
    }
    __syncthreads();

    // ---- Stage 2c: score = sum_n prb * (W@Xp) * tanh(V@Xp) ----
    for (int t = 0; t < (ROWS + 15) / 16; ++t) {
        v8f wx = {}, vx = {};
        wx = wmma_bf16(frag_pre(Wf, t * 2),     xb0, wx);
        wx = wmma_bf16(frag_pre(Wf, t * 2 + 1), xb1, wx);
        vx = wmma_bf16(frag_pre(Vf, t * 2),     xb0, vx);
        vx = wmma_bf16(frag_pre(Vf, t * 2 + 1), xb1, vx);
#pragma unroll
        for (int r = 0; r < 8; ++r) {
            int row = t * 16 + r + half * 8;                // row of W/V = n*C + c
            if (row < ROWS) {
                int n = row / C_DIM;
                int c = row - n * C_DIM;
                float s = prb[wave][n][ncol] * wx[r] * tanhf(vx[r]);
                atomicAdd(&sco[wave][c][ncol], s);          // ds_add_f32
            }
        }
    }
    __syncthreads();

    if (lane < 16) {
        int b = b0 + wave * 16 + lane;
#pragma unroll
        for (int c = 0; c < C_DIM; ++c)
            out[(size_t)c * B_DIM + b] = sco[wave][c][lane];
    }
}

extern "C" void kernel_launch(void* const* d_in, const int* in_sizes, int n_in,
                              void* d_out, int out_size, void* d_ws, size_t ws_size,
                              hipStream_t stream) {
    const float* X = (const float*)d_in[0];   // (B, D)
    const float* Z = (const float*)d_in[1];   // (P, D)
    const float* W = (const float*)d_in[2];   // (C*TOT, P)
    const float* V = (const float*)d_in[3];   // (C*TOT, P)
    const float* T = (const float*)d_in[4];   // (INT, P)
    float* out = (float*)d_out;               // (C, B)

    unsigned* ws = (unsigned*)d_ws;           // ~292 KB of pre-packed fragments
    unsigned* Zf = ws + ZF_OFF;
    unsigned* Wf = ws + WF_OFF;
    unsigned* Vf = ws + VF_OFF;
    unsigned* Tf = ws + TF_OFF;

    // Pre-pack constant operands into WMMA fragment layout (bf16), once per call.
    {
        int n, blk = 256;
        n = ZF_FRAGS * 256;
        hipLaunchKernelGGL(pack_b_frags, dim3((n + blk - 1) / blk), dim3(blk), 0, stream,
                           Z, Zf, D_DIM, 4, 32);
        n = WF_FRAGS * 256;
        hipLaunchKernelGGL(pack_a_frags, dim3((n + blk - 1) / blk), dim3(blk), 0, stream,
                           W, Wf, ROWS, P_DIM, 40, 2);
        hipLaunchKernelGGL(pack_a_frags, dim3((n + blk - 1) / blk), dim3(blk), 0, stream,
                           V, Vf, ROWS, P_DIM, 40, 2);
        n = TF_FRAGS * 256;
        hipLaunchKernelGGL(pack_a_frags, dim3((n + blk - 1) / blk), dim3(blk), 0, stream,
                           T, Tf, INT_N, P_DIM, 2, 2);
    }

    dim3 grid(B_DIM / COLS_WG);               // 1024 workgroups
    dim3 block(32 * NW);                      // 128 threads = 4 waves (wave32)
    hipLaunchKernelGGL(bonsai_fused_kernel, grid, block, 0, stream,
                       X, Zf, Wf, Vf, Tf, out);
}
